// Attention_67834713473655
// MI455X (gfx1250) — matrix operand order repaired and checked by reading, not compiled, not run on hardware
//
#include <hip/hip_runtime.h>

#define E_DIM 1024
#define A_DIM 512
#define D_DIM 1024
#define B_SZ  32
#define S_SZ  2048
#define NEGV  (-1e10f)

typedef __attribute__((ext_vector_type(16))) __bf16        v16bf;
typedef __attribute__((ext_vector_type(8)))  float         v8f;
typedef __attribute__((ext_vector_type(16))) unsigned short v16u;
typedef __attribute__((ext_vector_type(8)))  unsigned short v8u;
typedef __attribute__((ext_vector_type(4)))  unsigned short v4u;

__device__ __forceinline__ unsigned short f2bf(float f) {
  unsigned u = __builtin_bit_cast(unsigned, f);
  u += 0x7FFFu + ((u >> 16) & 1u);   // round-to-nearest-even
  return (unsigned short)(u >> 16);
}

// ---- Kernel 1: We_w (E x A, f32) -> WeT (A x E, bf16), column-major for B frags
__global__ __launch_bounds__(256) void wet_transpose_kernel(
    const float* __restrict__ We, unsigned short* __restrict__ WeT) {
  int idx = blockIdx.x * 256 + threadIdx.x;     // over E*A, coalesced read
  int e = idx >> 9;
  int a = idx & (A_DIM - 1);
  WeT[(size_t)a * E_DIM + e] = f2bf(We[idx]);
}

// ---- Kernel 2: comb[b][a] = dh[b]@Wd_w[:,a] + Wd_b[a] + We_b[a]
__global__ __launch_bounds__(256) void comb_kernel(
    const float* __restrict__ dh, const float* __restrict__ Wd,
    const float* __restrict__ Wdb, const float* __restrict__ Web,
    float* __restrict__ comb) {
  int idx = blockIdx.x * 256 + threadIdx.x;     // over B*A
  int b = idx >> 9;
  int a = idx & (A_DIM - 1);
  const float* h = dh + (size_t)b * D_DIM;
  float s = 0.f;
#pragma unroll 8
  for (int e = 0; e < D_DIM; ++e) s = fmaf(h[e], Wd[(size_t)e * A_DIM + a], s);
  comb[idx] = s + Wdb[a] + Web[a];
}

// ---- Kernel 3: fused enc-projection GEMM (bf16 WMMA) + tanh + dot(Wf) + mask
// grid (S/64, B), 256 threads = 8 waves. Block tile: 64 rows x 512 cols.
// Each wave owns ALL 4 M-tiles and 4 N-tiles (cols wv*64..+63):
//   -> every B fragment feeds 4 wmma, every A fragment feeds 4 wmma.
// LDS A-tile (64x32, bf16) is double-buffered: one barrier per K-step.
__global__ __launch_bounds__(256) void scores_kernel(
    const float* __restrict__ enc, const int* __restrict__ mask,
    const unsigned short* __restrict__ WeT, const float* __restrict__ comb,
    const float* __restrict__ Wf, const float* __restrict__ Wfb,
    float* __restrict__ scores) {
  __shared__ unsigned short Atile[2][64 * 32];  // 2 x 4 KB
  __shared__ float att_s[64];

  const int b    = blockIdx.y;
  const int s0   = blockIdx.x * 64;
  const int t    = threadIdx.x;
  const int lane = t & 31;
  const int wv   = t >> 5;
  const int nBase = wv * 64;            // 4 n-tiles of 16
  const int l15   = lane & 15;
  const int kg    = lane >> 4;

  if (t < 64) att_s[t] = 0.f;

  const float* encB = enc + ((size_t)b * S_SZ + s0) * E_DIM;

  v8f acc[4][4];                         // [m][nt]
#pragma unroll
  for (int m = 0; m < 4; ++m)
#pragma unroll
    for (int nt = 0; nt < 4; ++nt)
#pragma unroll
      for (int j = 0; j < 8; ++j) acc[m][nt][j] = 0.f;

  // staging helper: 64 rows x 32 K floats -> bf16 LDS buffer p
  auto stage = [&](int kk, int p) {
    const int k0 = kk * 32;
#pragma unroll
    for (int r = 0; r < 2; ++r) {
      const int idx = t + r * 256;       // 0..511 float4 slots
      const int row = idx >> 3;          // 8 float4 per 32-col row
      const int c4  = (idx & 7) * 4;
      const float4 f = *(const float4*)(encB + (size_t)row * E_DIM + k0 + c4);
      v4u h;
      h.x = f2bf(f.x); h.y = f2bf(f.y); h.z = f2bf(f.z); h.w = f2bf(f.w);
      *(v4u*)(&Atile[p][row * 32 + c4]) = h;
    }
  };

  stage(0, 0);

  for (int kk = 0; kk < 32; ++kk) {
    const int p  = kk & 1;
    const int k0 = kk * 32;
    __syncthreads();                     // buffer p staged; buffer p^1 free
    if (kk + 1 < 32) stage(kk + 1, p ^ 1);

    // A fragments (contiguous K layout: lane<16 -> K 0-15, lane>=16 -> K 16-31)
    v16u afrag[4];
#pragma unroll
    for (int m = 0; m < 4; ++m) {
      const int arow = m * 16 + l15;
      const v8u lo = *(const v8u*)(&Atile[p][arow * 32 + kg * 16]);
      const v8u hi = *(const v8u*)(&Atile[p][arow * 32 + kg * 16 + 8]);
#pragma unroll
      for (int i = 0; i < 8; ++i) { afrag[m][i] = lo[i]; afrag[m][i + 8] = hi[i]; }
    }

#pragma unroll
    for (int nt = 0; nt < 4; ++nt) {
      const int col = nBase + nt * 16 + l15;
      // B fragment: column 'col', K = k0 + kg*16 .. +15, contiguous in WeT
      const v16u bfrag = *(const v16u*)(WeT + (size_t)col * E_DIM + k0 + kg * 16);
#pragma unroll
      for (int m = 0; m < 4; ++m) {
        acc[m][nt] = __builtin_amdgcn_wmma_f32_16x16x32_bf16(
            false, __builtin_bit_cast(v16bf, afrag[m]),
            false, __builtin_bit_cast(v16bf, bfrag),
            (short)0, acc[m][nt], false, false);
      }
    }
  }

  // Epilogue: D layout is lane=N, VGPR v = row v+8*kg. att += tanh(enc+comb)*Wf.
  float att8[4][8];
#pragma unroll
  for (int m = 0; m < 4; ++m)
#pragma unroll
    for (int v = 0; v < 8; ++v) att8[m][v] = 0.f;

  const float* combB = comb + (size_t)b * A_DIM;
#pragma unroll
  for (int nt = 0; nt < 4; ++nt) {
    const int col = nBase + nt * 16 + l15;
    const float cb = combB[col];
    const float wf = Wf[col];
#pragma unroll
    for (int m = 0; m < 4; ++m)
#pragma unroll
      for (int v = 0; v < 8; ++v)
        att8[m][v] += tanhf(acc[m][nt][v] + cb) * wf;
  }
  // reduce across the 16 columns held by lanes (stays within 16-lane halves)
#pragma unroll
  for (int off = 1; off < 16; off <<= 1)
#pragma unroll
    for (int m = 0; m < 4; ++m)
#pragma unroll
      for (int v = 0; v < 8; ++v)
        att8[m][v] += __shfl_xor(att8[m][v], off, 32);

  if (l15 == 0) {
#pragma unroll
    for (int m = 0; m < 4; ++m)
#pragma unroll
      for (int v = 0; v < 8; ++v)
        atomicAdd(&att_s[m * 16 + kg * 8 + v], att8[m][v]);  // merge the 8 waves' column slices
  }
  __syncthreads();
  if (t < 64) {
    const int s = s0 + t;
    float sc = att_s[t] + Wfb[0];
    if (mask[(size_t)b * S_SZ + s] == 0) sc = NEGV;
    scores[(size_t)b * S_SZ + s] = sc;
  }
}

// ---- Kernel 4: masked softmax over S per batch row -> attention weights
__global__ __launch_bounds__(256) void softmax_kernel(
    const float* __restrict__ scores, float* __restrict__ wout) {
  __shared__ float red[256];
  const int b = blockIdx.x, t = threadIdx.x;
  const float* row = scores + (size_t)b * S_SZ;
  float v[8];
  float mx = -3.0e38f;
#pragma unroll
  for (int i = 0; i < 8; ++i) { v[i] = row[t + i * 256]; mx = fmaxf(mx, v[i]); }
  red[t] = mx; __syncthreads();
  for (int s = 128; s > 0; s >>= 1) {
    if (t < s) red[t] = fmaxf(red[t], red[t + s]);
    __syncthreads();
  }
  mx = red[0]; __syncthreads();
  float sum = 0.f;
#pragma unroll
  for (int i = 0; i < 8; ++i) { v[i] = __expf(v[i] - mx); sum += v[i]; }
  red[t] = sum; __syncthreads();
  for (int s = 128; s > 0; s >>= 1) {
    if (t < s) red[t] += red[t + s];
    __syncthreads();
  }
  const float inv = 1.f / red[0];
#pragma unroll
  for (int i = 0; i < 8; ++i) wout[(size_t)b * S_SZ + t + i * 256] = v[i] * inv;
}

// ---- Kernel 5: context[b][e] = sum_s w[b][s] * enc[b][s][e]  (bandwidth bound)
__global__ __launch_bounds__(128) void context_kernel(
    const float* __restrict__ enc, const float* __restrict__ w,
    float* __restrict__ ctx) {
  const int b = blockIdx.x;
  const int e = blockIdx.y * 128 + threadIdx.x;
  const float* eb = enc + (size_t)b * S_SZ * E_DIM + e;
  const float* wb = w + (size_t)b * S_SZ;
  float a0 = 0.f, a1 = 0.f, a2 = 0.f, a3 = 0.f;
  for (int s = 0; s < S_SZ; s += 4) {
    a0 = fmaf(wb[s + 0], eb[(size_t)(s + 0) * E_DIM], a0);
    a1 = fmaf(wb[s + 1], eb[(size_t)(s + 1) * E_DIM], a1);
    a2 = fmaf(wb[s + 2], eb[(size_t)(s + 2) * E_DIM], a2);
    a3 = fmaf(wb[s + 3], eb[(size_t)(s + 3) * E_DIM], a3);
  }
  ctx[(size_t)b * E_DIM + e] = (a0 + a1) + (a2 + a3);
}

extern "C" void kernel_launch(void* const* d_in, const int* in_sizes, int n_in,
                              void* d_out, int out_size, void* d_ws, size_t ws_size,
                              hipStream_t stream) {
  const float* enc  = (const float*)d_in[0];   // (B,S,E)
  const float* dh   = (const float*)d_in[1];   // (B,D)
  const int*   mask = (const int*)d_in[2];     // (B,S)
  const float* We_w = (const float*)d_in[3];   // (E,A)
  const float* We_b = (const float*)d_in[4];   // (A)
  const float* Wd_w = (const float*)d_in[5];   // (D,A)
  const float* Wd_b = (const float*)d_in[6];   // (A)
  const float* Wf_w = (const float*)d_in[7];   // (A)
  const float* Wf_b = (const float*)d_in[8];   // scalar

  float* wout = (float*)d_out;                 // (B,S) attention weights
  float* ctx  = wout + (size_t)B_SZ * S_SZ;    // (B,E) context

  char* ws = (char*)d_ws;
  unsigned short* WeT = (unsigned short*)ws;                       // 1 MB bf16
  float* comb   = (float*)(ws + (size_t)E_DIM * A_DIM * 2);        // 64 KB
  float* scores = (float*)(ws + (size_t)E_DIM * A_DIM * 2 +
                                (size_t)B_SZ * A_DIM * 4);         // 256 KB

  wet_transpose_kernel<<<(E_DIM * A_DIM) / 256, 256, 0, stream>>>(We_w, WeT);
  comb_kernel<<<(B_SZ * A_DIM) / 256, 256, 0, stream>>>(dh, Wd_w, Wd_b, We_b, comb);
  scores_kernel<<<dim3(S_SZ / 64, B_SZ), 256, 0, stream>>>(enc, mask, WeT, comb,
                                                           Wf_w, Wf_b, scores);
  softmax_kernel<<<B_SZ, 256, 0, stream>>>(scores, wout);
  context_kernel<<<dim3(B_SZ, E_DIM / 128), 128, 0, stream>>>(enc, wout, ctx);
}
